// XLMRobertaSelfAttention_24962349924459
// MI455X (gfx1250) — matrix operand order repaired and checked by reading, not compiled, not run on hardware
//
#include <hip/hip_runtime.h>

// ---------------- types / helpers ----------------
typedef __attribute__((ext_vector_type(16))) __bf16 v16bf;
typedef __attribute__((ext_vector_type(16))) unsigned short v16us;
typedef __attribute__((ext_vector_type(8))) float v8f;

#define S_LEN 512
#define HD 64
#define NH 16
#define NB 16
#define HID_ 1024
#define N3 3072

typedef unsigned long long u64t;

__device__ __forceinline__ unsigned short f2bf(float f) {
  union { float f; unsigned u; } x; x.f = f;
  unsigned r = x.u + 0x7FFFu + ((x.u >> 16) & 1u);   // round-to-nearest-even
  return (unsigned short)(r >> 16);
}
__device__ __forceinline__ v16bf bfc(v16us v) {
  union { v16us u; v16bf b; } x; x.u = v; return x.b;
}
__device__ __forceinline__ v8f wmma_bf16(v16bf a, v16bf b, v8f c) {
  return __builtin_amdgcn_wmma_f32_16x16x32_bf16(false, a, false, b, (short)0, c,
                                                 false, false);
}

// CDNA5 async DMA: global -> LDS, 16B per lane, tracked by ASYNCcnt.
__device__ __forceinline__ void async_b128(unsigned short* l,
                                           const unsigned short* g) {
  unsigned loff = (unsigned)(size_t)l;  // low 32 bits of generic = LDS offset
  asm volatile("global_load_async_to_lds_b128 %0, %1, off"
               :: "v"(loff), "v"(g) : "memory");
}
__device__ __forceinline__ void wait_async0() {
  asm volatile("s_wait_asynccnt 0" ::: "memory");
}

// A-frag (16x32 MxK slice) from row-major [M][K] LDS tile; one contiguous
// 16-element run per lane -> 2x ds_load_b128.
__device__ __forceinline__ v16bf a_frag_lds(const unsigned short* p0, int row0,
                                            int k0, int ld, int lane) {
  const unsigned short* p =
      p0 + (row0 + (lane & 15)) * ld + k0 + ((lane >> 4) << 4);
  v16us t;
#pragma unroll
  for (int e = 0; e < 16; ++e) t[e] = p[e];
  return bfc(t);
}
// B-frag where B[k][n] = M[row0+n][k] for row-major M tile (K^T, W^T, V^T):
// one contiguous 16-element run per lane -> 2x ds_load_b128.
__device__ __forceinline__ v16bf bt_frag_lds(const unsigned short* p0, int row0,
                                             int k0, int ld, int lane) {
  const unsigned short* p =
      p0 + (row0 + (lane & 15)) * ld + k0 + ((lane >> 4) << 4);
  v16us t;
#pragma unroll
  for (int e = 0; e < 16; ++e) t[e] = p[e];
  return bfc(t);
}
// PE B-frag from global bf16: B[k=d][p] = pe[(d0+pbase+p)*64 + k0 + d']
__device__ __forceinline__ v16bf pe_frag(const unsigned short* __restrict__ peg,
                                         int d0, int pbase, int k0, int lane) {
  const unsigned short* p =
      peg + (d0 + pbase + (lane & 15)) * 64 + k0 + ((lane >> 4) << 4);
  v16us t;
#pragma unroll
  for (int e = 0; e < 16; ++e) t[e] = p[e];
  return bfc(t);
}

#define LDT 72  // padded bf16 LDS row stride (144B: 16B aligned, no bank repeat)

// ---------------- kernel 0a: hidden f32 -> bf16 (same layout) ----------------
__global__ __launch_bounds__(256) void h_convert_kernel(
    const float* __restrict__ hid, unsigned short* __restrict__ Hb) {
  size_t i = (size_t)blockIdx.x * 256 + threadIdx.x;  // one float4 per thread
  const float4 v = *(const float4*)(hid + i * 4);
  unsigned lo = (unsigned)f2bf(v.x) | ((unsigned)f2bf(v.y) << 16);
  unsigned hi = (unsigned)f2bf(v.z) | ((unsigned)f2bf(v.w) << 16);
  *(uint2*)(Hb + i * 4) = make_uint2(lo, hi);
}

// ---------------- kernel 0b: W transpose + bf16 convert ----------------
// Wt[n][k] = bf16(W[k][n]); enables contiguous-per-lane B-fragment loads.
__global__ __launch_bounds__(256) void wt_convert_kernel(
    const float* __restrict__ W, unsigned short* __restrict__ Wt) {
  __shared__ unsigned short tile[64 * 68];
  const int t = threadIdx.x;
  const int k0 = blockIdx.x * 64;
  const int n0 = blockIdx.y * 64;
#pragma unroll
  for (int i = 0; i < 4; ++i) {  // coalesced f32 reads along n
    int f = i * 256 + t;
    int r = f >> 4, c = (f & 15) << 2;
    const float4 v = *(const float4*)(W + (size_t)(k0 + r) * N3 + n0 + c);
    tile[r * 68 + c + 0] = f2bf(v.x);
    tile[r * 68 + c + 1] = f2bf(v.y);
    tile[r * 68 + c + 2] = f2bf(v.z);
    tile[r * 68 + c + 3] = f2bf(v.w);
  }
  __syncthreads();
#pragma unroll
  for (int i = 0; i < 4; ++i) {  // coalesced u64 writes along k
    int f = i * 256 + t;
    int n = f >> 4, c4 = (f & 15) << 2;
    u64t v = (u64t)tile[(c4 + 0) * 68 + n]
           | ((u64t)tile[(c4 + 1) * 68 + n] << 16)
           | ((u64t)tile[(c4 + 2) * 68 + n] << 32)
           | ((u64t)tile[(c4 + 3) * 68 + n] << 48);
    *(u64t*)(Wt + (size_t)(n0 + n) * HID_ + k0 + c4) = v;
  }
}

// ---------------- kernel 1: fused QKV GEMM (async-DMA double-buffered) -------
// C[8192,3072] = Hb[8192,1024] @ W + b.  Q,K out [B,NH,S,HD]; V out transposed
// [B,NH,HD,S] (free in the scatter epilogue; contiguous B-frags for P@V).
#define BM 128
#define BN 128
#define BK 64
#define GEMM_LDS (2 * (BM * LDT + BN * LDT) * 2)  // 73728 B, double buffered

__global__ __launch_bounds__(256) void qkv_gemm_kernel(
    const unsigned short* __restrict__ Hb, const unsigned short* __restrict__ Wt,
    const float* __restrict__ bias, unsigned short* __restrict__ Qo,
    unsigned short* __restrict__ Ko, unsigned short* __restrict__ Vto) {
  extern __shared__ char gsm[];
  unsigned short* As = (unsigned short*)gsm;                      // 2x [128][72]
  unsigned short* Bs = (unsigned short*)(gsm + 2 * BM * LDT * 2); // 2x [128][72]
  const int t = threadIdx.x;
  const int lane = t & 31;
  const int w = t >> 5;
  const int wm = (w >> 2) * 64;  // wave sub-tile: 64 rows x 32 cols
  const int wn = (w & 3) * 32;
  const int m0 = blockIdx.y * BM;
  const int n0 = blockIdx.x * BN;

  v8f acc[4][2] = {};

  // stage(buf, kb): raw bf16 async copies, 4 b128 per thread per matrix
  auto stage = [&](int buf, int kb) {
    unsigned short* Ab = As + buf * BM * LDT;
    unsigned short* Bb = Bs + buf * BN * LDT;
#pragma unroll
    for (int i = 0; i < 4; ++i) {
      int f = i * 256 + t;
      int r = f >> 3, c8 = (f & 7) << 3;
      async_b128(Ab + r * LDT + c8, Hb + (size_t)(m0 + r) * HID_ + kb + c8);
      async_b128(Bb + r * LDT + c8, Wt + (size_t)(n0 + r) * HID_ + kb + c8);
    }
  };

  stage(0, 0);
  wait_async0();
  __syncthreads();

  for (int kb = 0; kb < HID_; kb += BK) {
    const int cur = (kb >> 6) & 1;
    if (kb + BK < HID_) stage(cur ^ 1, kb + BK);  // prefetch next tile
    const unsigned short* Ab = As + cur * BM * LDT;
    const unsigned short* Bb = Bs + cur * BN * LDT;
#pragma unroll
    for (int ks = 0; ks < BK; ks += 32) {
      v16bf b0 = bt_frag_lds(Bb, wn, ks, LDT, lane);
      v16bf b1 = bt_frag_lds(Bb, wn + 16, ks, LDT, lane);
#pragma unroll
      for (int i = 0; i < 4; ++i) {
        v16bf a = a_frag_lds(Ab, wm + i * 16, ks, LDT, lane);
        acc[i][0] = wmma_bf16(a, b0, acc[i][0]);
        acc[i][1] = wmma_bf16(a, b1, acc[i][1]);
      }
    }
    wait_async0();
    __syncthreads();
  }

  const int kh = lane >> 4;
#pragma unroll
  for (int i = 0; i < 4; ++i) {
#pragma unroll
    for (int j = 0; j < 2; ++j) {
      int n = n0 + wn + j * 16 + (lane & 15);
      float bv = bias[n];
      int seg = n / HID_;          // 0=Q 1=K 2=V (wave-uniform)
      int hh = (n % HID_) >> 6;
      int d = n & 63;
#pragma unroll
      for (int vi = 0; vi < 8; ++vi) {
        int m = m0 + wm + i * 16 + vi + (kh << 3);
        int bb = m >> 9;
        int s = m & 511;
        unsigned short val = f2bf(acc[i][j][vi] + bv);
        if (seg == 2) {  // V transposed: [b,h][d][s]
          Vto[((size_t)(bb * NH + hh) * HD + d) * S_LEN + s] = val;
        } else {
          unsigned short* dst = (seg == 0) ? Qo : Ko;
          dst[((size_t)(bb * NH + hh) * S_LEN + s) * HD + d] = val;
        }
      }
    }
  }
}

// ---------------- kernel 2: dist_emb f32 -> bf16 ----------------
__global__ void pe_convert_kernel(const float* __restrict__ pe,
                                  unsigned short* __restrict__ peo, int n) {
  int i = blockIdx.x * 256 + threadIdx.x;
  if (i < n) peo[i] = f2bf(pe[i]);
}

// ---------------- kernel 3: fused attention ----------------
// One block per (b, h, 64-row l-block). ~253KB dynamic LDS (CDNA5: 320KB/WGP).
#define SCP 516    // f32 stride of a score row
#define PBP 1032   // ushort stride of a prob row (aliases score row bytes)
#define VLD 520    // ushort stride of transposed-V LDS rows (1040B, 16B align)

#define OFF_VT (64 * SCP * 4)                 // 132096
#define OFF_QT (OFF_VT + HD * VLD * 2)        // +66560 -> 198656
#define OFF_KT (OFF_QT + 64 * LDT * 2)        // +9216  -> 207872 (2 buffers)
#define OFF_T1 (OFF_KT + 2 * 64 * LDT * 2)    // +18432 -> 226304
#define OFF_T2 (OFF_T1 + 8 * 512 * 4)         // +16384 -> 242688
#define ATTN_LDS (OFF_T2 + 8 * 512 * 4)       // 259072 bytes total

__global__ __launch_bounds__(256) void attn_kernel(
    const unsigned short* __restrict__ Qg, const unsigned short* __restrict__ Kg,
    const unsigned short* __restrict__ Vtg, const unsigned short* __restrict__ PEg,
    const float* __restrict__ mask, float* __restrict__ out) {
  extern __shared__ char smem[];
  float* sc = (float*)smem;                               // 64 x 516 f32 scores
  unsigned short* vtT = (unsigned short*)(smem + OFF_VT); // 64(d) x 520(r) bf16
  unsigned short* qt = (unsigned short*)(smem + OFF_QT);  // 64 x 72 bf16
  unsigned short* kt = (unsigned short*)(smem + OFF_KT);  // 2x 64 x 72 bf16
  float* t1b = (float*)(smem + OFF_T1);                   // per-wave 16x32
  float* t2b = (float*)(smem + OFF_T2);                   // per-wave 16x32

  const int t = threadIdx.x;
  const int lane = t & 31;
  const int w = t >> 5;
  const int kh = lane >> 4;
  const int l0 = blockIdx.x * 64;
  const int h = blockIdx.y;
  const int b = blockIdx.z;
  const size_t headoff = (size_t)(b * NH + h) * S_LEN * HD;
  const unsigned short* qsrc = Qg + headoff + (size_t)l0 * HD;
  const unsigned short* ksrc0 = Kg + headoff;
  const unsigned short* vsrc = Vtg + headoff;  // transposed: [d][s]

  // async-stage K chunk (into ping-pong buffer) + V chunk (disjoint vtT cols)
  auto stage_kv = [&](int r0, unsigned short* ktb) {
#pragma unroll
    for (int i = 0; i < 2; ++i) {
      int f = i * 256 + t;
      int r = f >> 3, c8 = (f & 7) << 3;
      async_b128(ktb + r * LDT + c8, ksrc0 + (size_t)(r0 + r) * HD + c8);
      async_b128(vtT + r * VLD + r0 + c8, vsrc + (size_t)r * S_LEN + r0 + c8);
    }
  };

  // prologue: Q block + chunk 0
#pragma unroll
  for (int i = 0; i < 2; ++i) {
    int f = i * 256 + t;
    int r = f >> 3, c8 = (f & 7) << 3;
    async_b128(qt + r * LDT + c8, qsrc + (size_t)r * HD + c8);
  }
  stage_kv(0, kt);
  wait_async0();
  __syncthreads();

  // ---- phase 1: scores + relative-position bias, streamed over 64-col chunks
  for (int c = 0; c < 8; ++c) {
    const int r0 = c * 64;
    const unsigned short* ktc = kt + (c & 1) * 64 * LDT;
    if (c < 7) stage_kv(r0 + 64, kt + ((c + 1) & 1) * 64 * LDT);  // prefetch

#pragma unroll
    for (int pp = 0; pp < 2; ++pp) {  // 16 tiles / 8 waves
      int tt = w + pp * 8;
      int li = tt >> 2, ri = tt & 3;
      int lt = li * 16, rt = ri * 16;
      // PE window base: dist = l - r + 511, tile window = d0 .. d0+30
      int d0 = (l0 + lt) - (r0 + rt) + 496;

      v8f sacc = {};
      v8f t1a[2] = {};
      v8f t2a[2] = {};
#pragma unroll
      for (int ksb = 0; ksb < HD; ksb += 32) {
        v16bf aq = a_frag_lds(qt, lt, ksb, LDT, lane);
        v16bf ak = a_frag_lds(ktc, rt, ksb, LDT, lane);
        v16bf bk = bt_frag_lds(ktc, rt, ksb, LDT, lane);
        sacc = wmma_bf16(aq, bk, sacc);  // Q . K^T
#pragma unroll
        for (int j = 0; j < 2; ++j) {    // T1 = Q.PEwin^T, T2 = K.PEwin^T
          v16bf pf = pe_frag(PEg, d0, j * 16, ksb, lane);
          t1a[j] = wmma_bf16(aq, pf, t1a[j]);
          t2a[j] = wmma_bf16(ak, pf, t2a[j]);
        }
      }
      float* t1w = t1b + w * 512;
      float* t2w = t2b + w * 512;
#pragma unroll
      for (int j = 0; j < 2; ++j)
#pragma unroll
        for (int vi = 0; vi < 8; ++vi) {
          t1w[(vi + (kh << 3)) * 32 + j * 16 + (lane & 15)] = t1a[j][vi];
          t2w[(vi + (kh << 3)) * 32 + j * 16 + (lane & 15)] = t2a[j][vi];
        }
      asm volatile("s_wait_dscnt 0" ::: "memory");  // order scratch write->gather
#pragma unroll
      for (int vi = 0; vi < 8; ++vi) {
        int lr = vi + (kh << 3);
        int rr = lane & 15;
        int p = lr - rr + 15;  // dist - d0, in [0,30]
        float v = sacc[vi] + t1w[lr * 32 + p] + t2w[rr * 32 + p];
        v = v * 0.125f + mask[b * S_LEN + r0 + rt + rr];  // /sqrt(64) + mask
        sc[(lt + lr) * SCP + (r0 + rt + rr)] = v;
      }
    }
    wait_async0();
    __syncthreads();
  }

  // ---- phase 2: softmax per row, re-store probs as bf16 in-place
  unsigned short* pb = (unsigned short*)sc;
  for (int r8 = 0; r8 < 8; ++r8) {
    int row = w * 8 + r8;
    float* srow = sc + row * SCP;
    float vals[16];
    float mx = -3.0e38f;
#pragma unroll
    for (int i = 0; i < 16; ++i) vals[i] = srow[lane + i * 32];
#pragma unroll
    for (int i = 0; i < 16; ++i) mx = fmaxf(mx, vals[i]);
#pragma unroll
    for (int off = 16; off > 0; off >>= 1) mx = fmaxf(mx, __shfl_xor(mx, off, 32));
    float sum = 0.0f;
#pragma unroll
    for (int i = 0; i < 16; ++i) { vals[i] = __expf(vals[i] - mx); sum += vals[i]; }
#pragma unroll
    for (int off = 16; off > 0; off >>= 1) sum += __shfl_xor(sum, off, 32);
    float inv = 1.0f / sum;  // every store below depends on all loads via sum
    unsigned short* prow = pb + row * PBP;
#pragma unroll
    for (int i = 0; i < 16; ++i) prow[lane + i * 32] = f2bf(vals[i] * inv);
  }
  __syncthreads();

  // ---- phase 3: ctx = P @ V   (B-frag contiguous from transposed V)
#pragma unroll
  for (int pp = 0; pp < 2; ++pp) {
    int tt = w + pp * 8;
    int li = tt >> 2, di = tt & 3;
    int lt = li * 16;
    v8f acc = {};
    for (int kr = 0; kr < S_LEN; kr += 32) {
      const unsigned short* p = pb + (lt + (lane & 15)) * PBP + kr + (kh << 4);
      v16us tr;
#pragma unroll
      for (int e = 0; e < 16; ++e) tr[e] = p[e];
      v16bf a = bfc(tr);
      v16bf bv = bt_frag_lds(vtT, di * 16, kr, VLD, lane);  // B[k=r][n=d]
      acc = wmma_bf16(a, bv, acc);
    }
#pragma unroll
    for (int vi = 0; vi < 8; ++vi) {
      int lg = l0 + lt + vi + (kh << 3);
      int d = di * 16 + (lane & 15);
      out[(size_t)(b * S_LEN + lg) * (NH * HD) + h * HD + d] = acc[vi];
    }
  }
}

// ---------------- launch ----------------
extern "C" void kernel_launch(void* const* d_in, const int* in_sizes, int n_in,
                              void* d_out, int out_size, void* d_ws,
                              size_t ws_size, hipStream_t stream) {
  const float* hid = (const float*)d_in[0];    // [16,512,1024]
  const float* mask = (const float*)d_in[1];   // [16,1,1,512]
  const float* W = (const float*)d_in[2];      // [1024,3072]
  const float* bqkv = (const float*)d_in[3];   // [3072]
  const float* pe = (const float*)d_in[4];     // [1023,64]
  float* out = (float*)d_out;                  // [16,512,1024]

  char* ws = (char*)d_ws;
  const size_t QKV_BYTES = (size_t)NB * NH * S_LEN * HD * 2;  // 16.78 MB each
  const size_t PE_BYTES = (size_t)1024 * 64 * 2;              // padded row 1023
  const size_t WT_BYTES = (size_t)N3 * HID_ * 2;
  unsigned short* Q = (unsigned short*)(ws);
  unsigned short* K = (unsigned short*)(ws + QKV_BYTES);
  unsigned short* Vt = (unsigned short*)(ws + 2 * QKV_BYTES);
  unsigned short* PE = (unsigned short*)(ws + 3 * QKV_BYTES);
  unsigned short* Wt = (unsigned short*)(ws + 3 * QKV_BYTES + PE_BYTES);
  unsigned short* Hb = (unsigned short*)(ws + 3 * QKV_BYTES + PE_BYTES + WT_BYTES);

  h_convert_kernel<<<(NB * S_LEN * HID_ / 4) / 256, 256, 0, stream>>>(hid, Hb);
  wt_convert_kernel<<<dim3(HID_ / 64, N3 / 64), 256, 0, stream>>>(W, Wt);

  const int npe = 1023 * 64;
  pe_convert_kernel<<<(npe + 255) / 256, 256, 0, stream>>>(pe, PE, npe);

  hipFuncSetAttribute((const void*)qkv_gemm_kernel,
                      hipFuncAttributeMaxDynamicSharedMemorySize, GEMM_LDS);
  qkv_gemm_kernel<<<dim3(N3 / BN, (NB * S_LEN) / BM), 256, GEMM_LDS, stream>>>(
      Hb, Wt, bqkv, Q, K, Vt);

  hipFuncSetAttribute((const void*)attn_kernel,
                      hipFuncAttributeMaxDynamicSharedMemorySize, ATTN_LDS);
  attn_kernel<<<dim3(S_LEN / 64, NH, NB), 256, ATTN_LDS, stream>>>(
      Q, K, Vt, PE, mask, out);
}
